// DistributedSelfAttention_3951369912664
// MI455X (gfx1250) — matrix operand order repaired and checked by reading, not compiled, not run on hardware
//
#include <hip/hip_runtime.h>
#include <hip/hip_bf16.h>

typedef __attribute__((ext_vector_type(16))) _Float16 v16h;
typedef __attribute__((ext_vector_type(8)))  _Float16 v8h;
typedef __attribute__((ext_vector_type(8)))  float    v8f;

constexpr int S   = 3072;
constexpr int HID = 1024;
constexpr int NH  = 16;
constexpr int HD  = 64;

__device__ __forceinline__ v8f wmma16(v16h a, v16h b, v8f c) {
    return __builtin_amdgcn_wmma_f32_16x16x32_f16(false, a, false, b, (short)0, c, false, false);
}

// A-fragment (16-bit, 16x32): lane<16 holds K={0..7,16..23}, lane>=16 K={8..15,24..31};
// caller folds the lane-half offset into p, so: e[0..7]=p[0..7], e[8..15]=p[16..23].
__device__ __forceinline__ v16h ldaf16(const _Float16* p) {
    v8h a = *(const v8h*)(p);
    v8h b = *(const v8h*)(p + 16);
    return __builtin_shufflevector(a, b, 0,1,2,3,4,5,6,7,8,9,10,11,12,13,14,15);
}

// A-fragment from f32 memory (converted to f16): e[0..7]=p[0..7], e[8..15]=p[16..23]
__device__ __forceinline__ v16h cvt_af32(const float* p) {
    const float4* q = (const float4*)p;
    float4 a = q[0], b = q[1], c = q[4], d = q[5];
    v16h r;
    r[0]=(_Float16)a.x;  r[1]=(_Float16)a.y;  r[2]=(_Float16)a.z;  r[3]=(_Float16)a.w;
    r[4]=(_Float16)b.x;  r[5]=(_Float16)b.y;  r[6]=(_Float16)b.z;  r[7]=(_Float16)b.w;
    r[8]=(_Float16)c.x;  r[9]=(_Float16)c.y;  r[10]=(_Float16)c.z; r[11]=(_Float16)c.w;
    r[12]=(_Float16)d.x; r[13]=(_Float16)d.y; r[14]=(_Float16)d.z; r[15]=(_Float16)d.w;
    return r;
}

// B-fragment from f32 memory: e[0..7]=p[0..7], e[8..15]=p[16..23] (lane-half offset folded by caller)
__device__ __forceinline__ v16h cvt_bf32(const float* p) {
    const float4* q = (const float4*)p;
    float4 a = q[0], b = q[1], c = q[4], d = q[5];
    v16h r;
    r[0]=(_Float16)a.x;  r[1]=(_Float16)a.y;  r[2]=(_Float16)a.z;  r[3]=(_Float16)a.w;
    r[4]=(_Float16)b.x;  r[5]=(_Float16)b.y;  r[6]=(_Float16)b.z;  r[7]=(_Float16)b.w;
    r[8]=(_Float16)c.x;  r[9]=(_Float16)c.y;  r[10]=(_Float16)c.z; r[11]=(_Float16)c.w;
    r[12]=(_Float16)d.x; r[13]=(_Float16)d.y; r[14]=(_Float16)d.z; r[15]=(_Float16)d.w;
    return r;
}

// ---------------- Kernel 1: QKV projection, f32 -> f16 outputs ----------------
// One wave computes a 16(row) x 64(one head) output strip of Q, K, or V.
// C = X @ W^T + b.  Q scaled by 1/sqrt(HD).  V written transposed: Vt[h][d][s].
__global__ void qkv_proj_f16(const float* __restrict__ X,
                             const float* __restrict__ Wq, const float* __restrict__ bq,
                             const float* __restrict__ Wk, const float* __restrict__ bk,
                             const float* __restrict__ Wv, const float* __restrict__ bv,
                             _Float16* __restrict__ Qo, _Float16* __restrict__ Ko,
                             _Float16* __restrict__ Vt) {
    const int  lane  = threadIdx.x;
    const bool lo    = lane < 16;
    const int  col   = lane & 15;
    const int  kbase = lo ? 0 : 8;    // A-fragment K offset for this lane half
    const int  kh    = lo ? 0 : 8;    // B-fragment K offset for this lane half
    const int  m0    = blockIdx.x * 16;
    const int  h     = blockIdx.y;
    const int  mat   = blockIdx.z;    // 0=Q, 1=K, 2=V
    const int  o0    = h * HD;

    const float* Wm = (mat == 0) ? Wq : ((mat == 1) ? Wk : Wv);
    const float* bm = (mat == 0) ? bq : ((mat == 1) ? bk : bv);

    v8f zero = 0.0f;
    v8f acc[4] = {zero, zero, zero, zero};

    const float* ap = X + (size_t)(m0 + col) * HID;
    for (int kk = 0; kk < HID; kk += 32) {
        v16h a = cvt_af32(ap + kk + kbase);
#pragma unroll
        for (int t = 0; t < 4; ++t) {
            // B[k][n] = W[o0+t*16+n][k] -> two 8-float runs of W row (+0..7, +16..23)
            const float* wr = Wm + (size_t)(o0 + t*16 + col) * HID + kk + kh;
            v16h b = cvt_bf32(wr);
            acc[t] = wmma16(a, b, acc[t]);
        }
    }

    if (mat == 2) {
        // V transposed: Vt[(o0 + d) * S + s]; lane's 8 rows are contiguous in s
#pragma unroll
        for (int t = 0; t < 4; ++t) {
            float bb = bm[o0 + t*16 + col];
            v8h vv;
#pragma unroll
            for (int r = 0; r < 8; ++r) vv[r] = (_Float16)(acc[t][r] + bb);
            _Float16* vp = Vt + (size_t)(o0 + t*16 + col) * S + m0 + kbase;
            *(v8h*)vp = vv;
        }
    } else {
        _Float16* Ob = (mat == 0) ? Qo : Ko;
        const float sc = (mat == 0) ? 0.125f : 1.0f;  // fold 1/sqrt(64) into Q
#pragma unroll
        for (int t = 0; t < 4; ++t) {
            float bb = bm[o0 + t*16 + col];
#pragma unroll
            for (int r = 0; r < 8; ++r) {
                int srow = m0 + kbase + r;
                Ob[(size_t)(h * S + srow) * HD + t*16 + col] =
                    (_Float16)((acc[t][r] + bb) * sc);
            }
        }
    }
}

// ---------------- Kernel 2: fused flash attention (one wave = 16 queries x 1 head) ----
// S^T = K @ Q^T (so every lane owns one query), online softmax, ctx^T = V^T @ P^T.
__global__ void attn_f16(const _Float16* __restrict__ Q, const _Float16* __restrict__ K,
                         const _Float16* __restrict__ Vt, const float* __restrict__ mask,
                         float* __restrict__ out) {
    const int  lane  = threadIdx.x;
    const bool lo    = lane < 16;
    const int  col   = lane & 15;     // this lane's query (and N-column) index
    const int  kbase = lo ? 0 : 8;
    const int  kh    = lo ? 0 : 16;
    const int  q0    = blockIdx.x * 16;
    const int  h     = blockIdx.y;

    // Q^T B-fragments, resident across the key loop (d-chunks 0..31 and 32..63)
    const _Float16* qrow = Q + (size_t)(h * S + q0 + col) * HD;
    v16h bq0 = ldaf16(qrow + kbase);
    v16h bq1 = ldaf16(qrow + 32 + kbase);

    const _Float16* Kh = K  + (size_t)h * S * HD;
    const _Float16* Vh = Vt + (size_t)h * HD * S;

    v8f zero = 0.0f;
    v8f acc[4] = {zero, zero, zero, zero};   // ctx^T: 64(d) x 16(q)
    float mrun = -3.0e38f;
    float lsum = 0.0f;

    for (int n0 = 0; n0 < S; n0 += 32) {
        // ---- scores (transposed): two 16-key tiles, K-dim = 64 (2 WMMAs each)
        const _Float16* k0 = Kh + (size_t)(n0 + col) * HD;
        const _Float16* k1 = Kh + (size_t)(n0 + 16 + col) * HD;
        v16h a00 = ldaf16(k0 + kbase);
        v16h a01 = ldaf16(k0 + 32 + kbase);
        v16h a10 = ldaf16(k1 + kbase);
        v16h a11 = ldaf16(k1 + 32 + kbase);
        v8f s0 = zero, s1 = zero;
        s0 = wmma16(a00, bq0, s0);  s0 = wmma16(a01, bq1, s0);
        s1 = wmma16(a10, bq0, s1);  s1 = wmma16(a11, bq1, s1);

        // ---- additive mask (per key = per acc row)
        const float4* mp0 = (const float4*)(mask + n0 + kbase);
        const float4* mp1 = (const float4*)(mask + n0 + 16 + kbase);
        float4 ma = mp0[0], mb = mp0[1], mc = mp1[0], md = mp1[1];
        s0[0]+=ma.x; s0[1]+=ma.y; s0[2]+=ma.z; s0[3]+=ma.w;
        s0[4]+=mb.x; s0[5]+=mb.y; s0[6]+=mb.z; s0[7]+=mb.w;
        s1[0]+=mc.x; s1[1]+=mc.y; s1[2]+=mc.z; s1[3]+=mc.w;
        s1[4]+=md.x; s1[5]+=md.y; s1[6]+=md.z; s1[7]+=md.w;

        // ---- online softmax; per-query stats live in lanes q and q+16
        float lm = fmaxf(s0[0], s1[0]);
#pragma unroll
        for (int r = 1; r < 8; ++r) lm = fmaxf(lm, fmaxf(s0[r], s1[r]));
        lm = fmaxf(lm, __shfl_xor(lm, 16, 32));
        float mnew = fmaxf(mrun, lm);
        float scl  = __expf(mrun - mnew);
        float p0[8], p1[8];
        float ls = 0.0f;
#pragma unroll
        for (int r = 0; r < 8; ++r) {
            p0[r] = __expf(s0[r] - mnew);
            p1[r] = __expf(s1[r] - mnew);
            ls += p0[r] + p1[r];
        }
        ls += __shfl_xor(ls, 16, 32);
        lsum = lsum * scl + ls;
        mrun = mnew;
#pragma unroll
        for (int t = 0; t < 4; ++t)
#pragma unroll
            for (int r = 0; r < 8; ++r) acc[t][r] *= scl;

        // ---- build P^T B-fragment: lane<16 needs keys {0..7,16..23}, lane>=16 keys {8..15,24..31}
        v16h pf;
#pragma unroll
        for (int j = 0; j < 8; ++j) {
            float o1 = __shfl_xor(p1[j], 16, 32);
            float o0 = __shfl_xor(p0[j], 16, 32);
            pf[j]     = (_Float16)p0[j];
            pf[j + 8] = (_Float16)p1[j];
        }

        // ---- ctx^T += V^T tile x P^T  (4 d-chunks of 16)
#pragma unroll
        for (int t = 0; t < 4; ++t) {
            const _Float16* vp = Vh + (size_t)(t*16 + col) * S + n0 + kbase;
            v16h av = ldaf16(vp);
            acc[t] = wmma16(av, pf, acc[t]);
        }
    }

    // ---- normalize + write out[s][h*64+d]; lane's 8 rows (r) are contiguous in d
    float rinv = (lsum > 0.0f) ? 1.0f / lsum : 0.0f;
    float* orow = out + (size_t)(q0 + col) * HID + h * HD;
#pragma unroll
    for (int t = 0; t < 4; ++t) {
        int d0 = t*16 + kbase;
        float4 w0 = make_float4(acc[t][0]*rinv, acc[t][1]*rinv, acc[t][2]*rinv, acc[t][3]*rinv);
        float4 w1 = make_float4(acc[t][4]*rinv, acc[t][5]*rinv, acc[t][6]*rinv, acc[t][7]*rinv);
        *(float4*)(orow + d0)     = w0;
        *(float4*)(orow + d0 + 4) = w1;
    }
}

extern "C" void kernel_launch(void* const* d_in, const int* in_sizes, int n_in,
                              void* d_out, int out_size, void* d_ws, size_t ws_size,
                              hipStream_t stream) {
    const float* X    = (const float*)d_in[0];
    const float* mask = (const float*)d_in[1];
    const float* Wq   = (const float*)d_in[2];
    const float* bq   = (const float*)d_in[3];
    const float* Wk   = (const float*)d_in[4];
    const float* bk   = (const float*)d_in[5];
    const float* Wv   = (const float*)d_in[6];
    const float* bv   = (const float*)d_in[7];
    float* out = (float*)d_out;

    _Float16* Qw = (_Float16*)d_ws;                 // [NH][S][HD] f16
    _Float16* Kw = Qw + (size_t)NH * S * HD;        // [NH][S][HD] f16
    _Float16* Vw = Kw + (size_t)NH * S * HD;        // [NH][HD][S] f16 (transposed)

    dim3 g1(S / 16, NH, 3);
    qkv_proj_f16<<<g1, 32, 0, stream>>>(X, Wq, bq, Wk, bk, Wv, bv, Qw, Kw, Vw);

    dim3 g2(S / 16, NH);
    attn_f16<<<g2, 32, 0, stream>>>(Qw, Kw, Vw, mask, out);

    (void)in_sizes; (void)n_in; (void)out_size; (void)ws_size;
}
